// LSTM_53815940218945
// MI455X (gfx1250) — matrix operand order repaired and checked by reading, not compiled, not run on hardware
//
#include <hip/hip_runtime.h>
#include <hip/hip_bf16.h>

typedef __attribute__((ext_vector_type(16))) _Float16 v16h;
typedef __attribute__((ext_vector_type(8)))  float    v8f;

#define HID   50
#define TLEN  256
#define BATCH 8192

// gfx1250 has a native V_TANH_F32 TRANS op; WMMA shares the TRANS tracking
// path, so 1-op activations matter. Guarded fallback keeps compile safe.
__device__ __forceinline__ float fast_tanh(float x) {
#if __has_builtin(__builtin_amdgcn_tanhf)
    return __builtin_amdgcn_tanhf(x);
#else
    float e = __builtin_amdgcn_exp2f(x * -2.8853900817779268f);
    return (1.0f - e) * __builtin_amdgcn_rcpf(1.0f + e);
#endif
}
__device__ __forceinline__ float fast_sigmoid(float x) {
    // sigmoid(x) = 0.5 + 0.5 * tanh(x/2)  -> exactly one TRANS op
    return fmaf(0.5f, fast_tanh(0.5f * x), 0.5f);
}

// One wave = 16 batch rows. Block = 4 waves = 64 batch rows.
// LDS holds Waug^T (K=64 x N=256) pre-swizzled into WMMA A-fragment layout:
//   32 fragments (nt=0..15 x kt=0..1), each fragment = 32 lanes x v16h.
// A-fragment element (lane, halfslot s): row n = nt*16 + lane%16,
//   K = kt*32 + s + (lane>=16?16:0)   (contiguous k order; B uses the same)
// The compiler hoists all 32 fragments into (high) VGPRs across the t-loop.
__global__ __launch_bounds__(128) void lstm_wmma_kernel(
    const float* __restrict__ x,     // [B, T, 1]
    const float* __restrict__ W_ih,  // [4H, 1]
    const float* __restrict__ W_hh,  // [4H, H]
    const float* __restrict__ b_ih,  // [4H]
    const float* __restrict__ b_hh,  // [4H]
    const float* __restrict__ W_lin, // [1, H]
    const float* __restrict__ b_lin, // [1]
    float* __restrict__ out)         // [B, 1]
{
    __shared__ alignas(32) _Float16 aFrag[32 * 32 * 16];  // 32 KB

    const int tid = threadIdx.x;

    // ---- Build augmented, padded, pre-swizzled weight fragments in LDS ----
    // Waug[k][n']: n' = g*64 + j (g=gate, j=hidden; j,K padded to 64)
    //   k <  50 : W_hh[g*50+j][k]
    //   k == 50 : W_ih[g*50+j]          (x slot; B carries x_t)
    //   k == 51 : b_ih[g*50+j]+b_hh[..] (bias slot; B carries 1.0)
    //   else 0
    for (int idx = tid; idx < 32 * 32; idx += 128) {
        const int f   = idx >> 5, l = idx & 31;
        const int nt  = f >> 1,  kt = f & 1;
        const int lhi = (l >> 4) & 1;
        const int n   = nt * 16 + (l & 15);
        const int g   = n >> 6, jj = n & 63;
        const int row = g * HID + jj;
        _Float16* dst = &aFrag[idx * 16];
        #pragma unroll
        for (int s = 0; s < 16; ++s) {
            const int k = kt * 32 + s + lhi * 16;
            float v = 0.0f;
            if (jj < HID) {
                if (k < HID)            v = W_hh[row * HID + k];
                else if (k == HID)      v = W_ih[row];
                else if (k == HID + 1)  v = b_ih[row] + b_hh[row];
            }
            dst[s] = (_Float16)v;
        }
    }
    __syncthreads();

    const int  wave = tid >> 5;
    const int  lane = tid & 31;
    const bool hi   = lane >= 16;
    const int  nl   = lane & 15;
    const int  b    = blockIdx.x * 64 + wave * 16 + nl;  // batch row (cols of D)
    const float* xrow = x + (long)b * TLEN;

    // cell / hidden state in D-layout: cst[jt][r] is j = jt*16 + r + (hi?8:0),
    // batch m = lane%16.
    float cst[4][8], hcur[4][8];
    #pragma unroll
    for (int jt = 0; jt < 4; ++jt)
        #pragma unroll
        for (int r = 0; r < 8; ++r) { cst[jt][r] = 0.0f; hcur[jt][r] = 0.0f; }

    #pragma unroll 1
    for (int t = 0; t < TLEN; ++t) {
        const float xv = xrow[t];

        // ---- Build B operand (haug^T, K=64 x M=16) from h in D-layout ----
        // B layout: lanes 0-15 hold K = s, lanes 16-31 hold K = 16+s (col m = lane%16)
        v16h B0, B1;
        #pragma unroll
        for (int r = 0; r < 8; ++r) {
            const float x0 = __shfl_xor(hcur[0][r], 16, 32);
            const float x1 = __shfl_xor(hcur[1][r], 16, 32);
            const float x2 = __shfl_xor(hcur[2][r], 16, 32);
            const float x3 = __shfl_xor(hcur[3][r], 16, 32);
            // K-tile 0: j = 0..31
            B0[r]     = (_Float16)(hi ? x1         : hcur[0][r]);
            B0[r + 8] = (_Float16)(hi ? hcur[1][r] : x0);
            // K-tile 1: j = 32..63 ; hi lanes carry j=48,49 then x, 1.0, zeros
            const float topv = (r < 2) ? x3 : ((r == 2) ? xv : ((r == 3) ? 1.0f : 0.0f));
            B1[r]     = (_Float16)(hi ? topv : hcur[2][r]);
            B1[r + 8] = (_Float16)(hi ? 0.0f : x2);
        }

        // ---- gates^T = Waug^T x haug^T, then fused cell update per j-tile ----
        #pragma unroll
        for (int jt = 0; jt < 4; ++jt) {
            v8f acc[4];  // i, f, g, o tiles for this j-tile
            #pragma unroll
            for (int gg = 0; gg < 4; ++gg) {
                const int nt = gg * 4 + jt;
                const v16h a0 = *(const v16h*)&aFrag[((nt * 2 + 0) * 32 + lane) * 16];
                const v16h a1 = *(const v16h*)&aFrag[((nt * 2 + 1) * 32 + lane) * 16];
                v8f z = {0.f, 0.f, 0.f, 0.f, 0.f, 0.f, 0.f, 0.f};
                z = __builtin_amdgcn_wmma_f32_16x16x32_f16(false, a0, false, B0,
                                                           (short)0, z, false, false);
                z = __builtin_amdgcn_wmma_f32_16x16x32_f16(false, a1, false, B1,
                                                           (short)0, z, false, false);
                acc[gg] = z;
            }
            #pragma unroll
            for (int r = 0; r < 8; ++r) {
                // jt==3, r>=2 covers j=50..55 (lo) / 58..63 (hi): pure padding in
                // BOTH lane halves; never read by B-build or the output dot.
                // Skip at compile time -> state stays exactly 0, fewer TRANS ops.
                if (jt == 3 && r >= 2) continue;
                const float iv = fast_sigmoid(acc[0][r]);
                const float fv = fast_sigmoid(acc[1][r]);
                const float gv = fast_tanh   (acc[2][r]);
                const float ov = fast_sigmoid(acc[3][r]);
                const float c  = fv * cst[jt][r] + iv * gv;
                cst[jt][r]  = c;
                hcur[jt][r] = ov * fast_tanh(c);
            }
        }
    }

    // ---- out[b] = h_T . W_lin + b_lin  (partial per lane-half, then xor-16 add)
    float partial = 0.0f;
    #pragma unroll
    for (int jt = 0; jt < 4; ++jt)
        #pragma unroll
        for (int r = 0; r < 8; ++r) {
            const int j = jt * 16 + r + (hi ? 8 : 0);
            const float wl = (j < HID) ? W_lin[j] : 0.0f;
            partial += hcur[jt][r] * wl;
        }
    partial += __shfl_xor(partial, 16, 32);
    if (lane < 16) out[b] = partial + b_lin[0];
}

extern "C" void kernel_launch(void* const* d_in, const int* in_sizes, int n_in,
                              void* d_out, int out_size, void* d_ws, size_t ws_size,
                              hipStream_t stream) {
    (void)in_sizes; (void)n_in; (void)out_size; (void)d_ws; (void)ws_size;
    const float* x     = (const float*)d_in[0];
    const float* W_ih  = (const float*)d_in[1];
    const float* W_hh  = (const float*)d_in[2];
    const float* b_ih  = (const float*)d_in[3];
    const float* b_hh  = (const float*)d_in[4];
    const float* W_lin = (const float*)d_in[5];
    const float* b_lin = (const float*)d_in[6];
    float* out = (float*)d_out;

    dim3 grid(BATCH / 64);   // 128 blocks, 64 batch rows each
    dim3 block(128);         // 4 wave32s
    lstm_wmma_kernel<<<grid, block, 0, stream>>>(x, W_ih, W_hh, b_ih, b_hh,
                                                 W_lin, b_lin, out);
}
